// PPR_34918084116721
// MI455X (gfx1250) — matrix operand order repaired and checked by reading, not compiled, not run on hardware
//
#include <hip/hip_runtime.h>
#include <hip/hip_bf16.h>
#include <stdint.h>

// ---------------------------------------------------------------------------
// PPR encode + weighted neighbor aggregation, tuned for MI455X (gfx1250).
//  enc = X @ W + b         (100000x1024 @ 1024x512, fp32 in/out)
//  out[b,:] = sum_k values[b,k] * enc[indices[b,k], :]
//
// GEMM uses v_wmma_f32_16x16x32_bf16 with a hi/lo bf16 split of both X and W
// (3 WMMAs per product tile) -> near-fp32 accuracy at ~3/8 of bf16 matrix
// peak, keeping the kernel memory-bound per the 23.3 TB/s roofline.
// X strip is staged into LDS with gfx1250 async copies
// (global_load_async_to_lds_b128 + s_wait_asynccnt).
// ---------------------------------------------------------------------------

typedef __attribute__((ext_vector_type(16))) __bf16 v16bf;
typedef __attribute__((ext_vector_type(8)))  float  v8f;
typedef __attribute__((ext_vector_type(2)))  float  v2f;

#define N_NODES 100000
#define D_IN    1024
#define D_OUT   512
#define BATCH   8192
#define TOPK    64

#define ROWS_PER_BLOCK 32      // M-tile per block (2 x 16-row WMMA tiles)
#define KSTAGE         256     // K columns staged in LDS at a time (4 stages)
#define LDS_STRIDE     260     // 256 + 4 pad -> bank step 4: conflict-free b128
                               // reads across 16 rows; 32*260*4 = 33280 B

#define USE_ASYNC_LDS 1        // gfx1250 async global->LDS copy path

// ------------------------- kernel 1: pack W --------------------------------
// Repack W[k][n] (row-major fp32) into WMMA B-fragment order, bf16 hi/lo:
//   packed[((nt*32 + kc)*32 + lane)*16 + e]
//   lane<16 :  n = nt*16+lane,    K = kc*32 + e        (e = 0..15)
//   lane>=16:  n = nt*16+lane-16, K = kc*32 + 16 + e
// so the GEMM loads one v16bf (32 contiguous bytes) per lane per fragment.
__global__ __launch_bounds__(256) void ppr_pack_w(
    const float* __restrict__ W, __bf16* __restrict__ whi, __bf16* __restrict__ wlo)
{
    int idx  = blockIdx.x * 256 + threadIdx.x;      // 0 .. 524287
    int e    = idx & 15;
    int lane = (idx >> 4) & 31;
    int kc   = (idx >> 9) & 31;
    int nt   = idx >> 14;
    int n    = nt * 16 + (lane & 15);
    int k    = kc * 32 + ((lane & 16) ? 16 : 0) + e;
    float w  = W[(size_t)k * D_OUT + n];
    __bf16 hi = (__bf16)w;
    whi[idx] = hi;
    wlo[idx] = (__bf16)(w - (float)hi);
}

// ------------------------- kernel 2: WMMA GEMM -----------------------------
// Block: 32-row strip of X, 256 threads = 8 waves; wave w covers cols
// [64w, 64w+64) for BOTH 16-row M-tiles (B fragments reused across M).
// Per 32-K chunk per wave: 2 A hi/lo fragments from LDS, 4 B hi/lo fragments
// from L2-resident packed W, 24 chained WMMAs with fp32 accumulation.
__global__ __launch_bounds__(256) void ppr_gemm_wmma(
    const float*  __restrict__ X,
    const __bf16* __restrict__ whi,
    const __bf16* __restrict__ wlo,
    const float*  __restrict__ bias,
    float*        __restrict__ enc)
{
    __shared__ float tileX[ROWS_PER_BLOCK * LDS_STRIDE];

    const int tid   = threadIdx.x;
    const int row0  = blockIdx.x * ROWS_PER_BLOCK;
    const int wave  = tid >> 5;
    const int lane  = tid & 31;
    const int lrow  = lane & 15;
    const int khalf = (lane & 16) ? 16 : 0;     // A-layout: lanes 16-31 at K+16
    const int nt0   = wave * 4;

    v8f acc[2][4];
#pragma unroll
    for (int mt = 0; mt < 2; ++mt)
#pragma unroll
        for (int t = 0; t < 4; ++t)
            acc[mt][t] = (v8f){0.f,0.f,0.f,0.f,0.f,0.f,0.f,0.f};

    // Cooperative fill geometry: thread -> (row = tid/8, 32-float slab).
    const int fr  = tid >> 3;
    const int fc0 = (tid & 7) * 32;

    for (int stage = 0; stage < 4; ++stage) {
        if (stage) __syncthreads();
        {
            const float* src = X + (size_t)(row0 + fr) * D_IN + stage * KSTAGE + fc0;
            float*       dst = tileX + fr * LDS_STRIDE + fc0;
#if USE_ASYNC_LDS
            // gfx1250 async copy: global -> LDS, no VGPR round-trip.
#pragma unroll
            for (int j = 0; j < 8; ++j) {
                unsigned           ldsa = (unsigned)(unsigned long long)(dst + j * 4);
                unsigned long long ga   = (unsigned long long)(src + j * 4);
                asm volatile("global_load_async_to_lds_b128 %0, %1, off"
                             :: "v"(ldsa), "v"(ga) : "memory");
            }
            asm volatile("s_wait_asynccnt 0" ::: "memory");
#else
#pragma unroll
            for (int j = 0; j < 8; ++j)
                ((float4*)dst)[j] = ((const float4*)src)[j];
#endif
        }
        __syncthreads();

        for (int kc = 0; kc < KSTAGE / 32; ++kc) {
            // ---- A fragments (contiguous k order, same as packed W): per lane 16 values:
            //      K = base+khalf+{0..15}
            v16bf ahi[2], alo[2];
#pragma unroll
            for (int mt = 0; mt < 2; ++mt) {
                const float* ap = tileX + (mt * 16 + lrow) * LDS_STRIDE
                                        + kc * 32 + khalf;
                float4 x0 = *(const float4*)(ap +  0);
                float4 x1 = *(const float4*)(ap +  4);
                float4 x2 = *(const float4*)(ap +  8);
                float4 x3 = *(const float4*)(ap + 12);
                float xf[16] = { x0.x,x0.y,x0.z,x0.w, x1.x,x1.y,x1.z,x1.w,
                                 x2.x,x2.y,x2.z,x2.w, x3.x,x3.y,x3.z,x3.w };
#pragma unroll
                for (int e = 0; e < 16; ++e) {
                    __bf16 hi = (__bf16)xf[e];
                    ahi[mt][e] = hi;
                    alo[mt][e] = (__bf16)(xf[e] - (float)hi);
                }
            }

            const int kcg = stage * (KSTAGE / 32) + kc;   // global 32-K chunk
#pragma unroll
            for (int t = 0; t < 4; ++t) {
                const size_t boff =
                    ((size_t)(((nt0 + t) * 32 + kcg) * 32 + lane)) * 16;
                v16bf bhi = *(const v16bf*)(whi + boff);
                v16bf blo = *(const v16bf*)(wlo + boff);
#pragma unroll
                for (int mt = 0; mt < 2; ++mt) {
                    // D = A*B + C ; fp32 accumulate; hi*hi + hi*lo + lo*hi
                    acc[mt][t] = __builtin_amdgcn_wmma_f32_16x16x32_bf16(
                        false, ahi[mt], false, bhi, (short)0, acc[mt][t], false, false);
                    acc[mt][t] = __builtin_amdgcn_wmma_f32_16x16x32_bf16(
                        false, ahi[mt], false, blo, (short)0, acc[mt][t], false, false);
                    acc[mt][t] = __builtin_amdgcn_wmma_f32_16x16x32_bf16(
                        false, alo[mt], false, bhi, (short)0, acc[mt][t], false, false);
                }
            }
        }
    }

    // ---- store with bias. C/D layout: VGPR i -> M=i (lanes 0-15) / M=8+i.
#pragma unroll
    for (int mt = 0; mt < 2; ++mt) {
#pragma unroll
        for (int t = 0; t < 4; ++t) {
            const int col = (nt0 + t) * 16 + lrow;
            const float bv = bias[col];
#pragma unroll
            for (int i = 0; i < 8; ++i) {
                const int m = mt * 16 + ((lane & 16) ? (8 + i) : i);
                enc[(size_t)(row0 + m) * D_OUT + col] = acc[mt][t][i] + bv;
            }
        }
    }
}

// ------------------------- kernel 3: weighted gather -----------------------
// One block per batch row; thread t owns columns {2t, 2t+1}. enc (205 MB) is
// ~L2-resident (192 MB), so the 1 GB of gather reads mostly hit L2.
__global__ __launch_bounds__(256) void ppr_gather(
    const int*   __restrict__ indices,
    const float* __restrict__ values,
    const float* __restrict__ enc,
    float*       __restrict__ out)
{
    __shared__ int   sidx[TOPK];
    __shared__ float sval[TOPK];
    const int b = blockIdx.x;
    const int t = threadIdx.x;
    if (t < TOPK) {
        sidx[t] = indices[(size_t)b * TOPK + t];
        sval[t] = values [(size_t)b * TOPK + t];
    }
    __syncthreads();

    const int col = t * 2;
    v2f c = (v2f){0.f, 0.f};
#pragma unroll 4
    for (int k = 0; k < TOPK; ++k) {
        const v2f e = *(const v2f*)(enc + (size_t)sidx[k] * D_OUT + col);
        const float v = sval[k];
        c.x = fmaf(v, e.x, c.x);
        c.y = fmaf(v, e.y, c.y);
    }
    // out is never re-read: bypass caches, keep L2 for enc.
    __builtin_nontemporal_store(c, (v2f*)(out + (size_t)b * D_OUT + col));
}

// ------------------------- launch ------------------------------------------
extern "C" void kernel_launch(void* const* d_in, const int* in_sizes, int n_in,
                              void* d_out, int out_size, void* d_ws, size_t ws_size,
                              hipStream_t stream) {
    const float* X       = (const float*)d_in[0];   // [100000,1024]
    const int*   indices = (const int*)  d_in[1];   // [8192,64]
    const float* values  = (const float*)d_in[2];   // [8192,64]
    const float* W       = (const float*)d_in[3];   // [1024,512]
    const float* bias    = (const float*)d_in[4];   // [512]
    float*       out     = (float*)d_out;           // [8192,512]

    // workspace layout: enc (204.8 MB) | Whi (1 MB) | Wlo (1 MB)
    const size_t ENC_BYTES = (size_t)N_NODES * D_OUT * sizeof(float); // 204800000
    const size_t WPACK     = (size_t)(D_IN / 32) * (D_OUT / 16) * 32 * 16; // 524288
    float*  enc = (float*)d_ws;
    __bf16* whi = (__bf16*)((char*)d_ws + ENC_BYTES);
    __bf16* wlo = whi + WPACK;

    ppr_pack_w   <<<(int)(WPACK / 256),       256, 0, stream>>>(W, whi, wlo);
    ppr_gemm_wmma<<<N_NODES / ROWS_PER_BLOCK, 256, 0, stream>>>(X, whi, wlo, bias, enc);
    ppr_gather   <<<BATCH,                    256, 0, stream>>>(indices, values, enc, out);
}
